// cross_set_score_77034533421806
// MI455X (gfx1250) — matrix operand order repaired and checked by reading, not compiled, not run on hardware
//
#include <hip/hip_runtime.h>
#include <hip/hip_bf16.h>

// ---- problem constants (from reference) ----
constexpr int kS   = 32;    // pair grid side
constexpr int kN   = 64;    // items per set
constexpr int kDin = 256;   // input feature dim
constexpr int kH   = 8;     // heads
constexpr int kD   = 64;    // head dim
constexpr int kHD  = kH * kD;          // 512
constexpr int kHNd = kH * kN * kD;     // 32768 f16 elements of h per pair

typedef __attribute__((ext_vector_type(16))) _Float16 v16h;
typedef __attribute__((ext_vector_type(8)))  _Float16 v8h;
typedef __attribute__((ext_vector_type(8)))  float    v8f;
typedef __attribute__((ext_vector_type(4)))  float    v4f;

union AFrag { v16h v; v8h h[2]; };

// ---------------------------------------------------------------------------
// Kernel 0: W [Din=256][HD=512] fp32  ->  Wt [HD=512][Din=256] f16
// Column-major copy so stage-1 B-fragments are 32B-contiguous loads.
// ---------------------------------------------------------------------------
__global__ __launch_bounds__(256) void wt_convert(const float* __restrict__ W,
                                                  _Float16* __restrict__ Wt) {
  int o = blockIdx.x * 256 + threadIdx.x;   // 0 .. 512*256-1
  int n = o >> 8;                           // output column of W  (0..511)
  int k = o & 255;                          // input feature       (0..255)
  Wt[o] = (_Float16)W[k * kHD + n];
}

// ---------------------------------------------------------------------------
// Kernel 1: per pair p=(j,i), compute h = x_p (64x256) @ W (256x512) in f16.
// x block staged in LDS as f16 (stride 264 halves to avoid bank conflicts).
// 8 waves x 16 tiles of 16x16, K looped in 8 chunks of 32 (wmma f16 k32).
// Output layout: h[p][head][n][d].
// ---------------------------------------------------------------------------
__global__ __launch_bounds__(256) void proj_gemm(const float* __restrict__ x,
                                                 const _Float16* __restrict__ Wt,
                                                 _Float16* __restrict__ hOut) {
  __shared__ _Float16 xs[kN * 264];
  const int p = blockIdx.x;                      // j*32 + i
  const float* xb = x + (size_t)p * (kN * kDin);

  // cooperative load + fp32->f16 convert: 16384 floats, float4 per step
  for (int t = 0; t < 16; ++t) {
    int i4  = threadIdx.x + t * 256;             // float4 index 0..4095
    v4f v   = *(const v4f*)(xb + (size_t)i4 * 4);
    int row = (i4 * 4) >> 8;
    int col = (i4 * 4) & 255;
    _Float16* dst = &xs[row * 264 + col];
    dst[0] = (_Float16)v.x; dst[1] = (_Float16)v.y;
    dst[2] = (_Float16)v.z; dst[3] = (_Float16)v.w;
  }
  __syncthreads();

  const int lane = threadIdx.x & 31;
  const int wave = threadIdx.x >> 5;
  const int m    = lane & 15;
  const int hi   = lane >> 4;

  for (int t = 0; t < 16; ++t) {
    int tile  = wave * 16 + t;                   // 0..127
    int nblk  = tile & 3;                        // row-tile   (0..3)
    int cblk  = tile >> 2;                       // col-tile   (0..31)
    int nbase = nblk * 16;
    int ccol  = cblk * 16 + m;                   // this lane's output column

    v8f acc = {};
    for (int kc = 0; kc < 8; ++kc) {
      // A fragment (16x32 f16): halves 0-7 -> K klo.., halves 8-15 -> K khi..
      int klo = kc * 32 + hi * 16;
      int khi = kc * 32 + hi * 16 + 8;
      AFrag a;
      a.h[0] = *(const v8h*)&xs[(nbase + m) * 264 + klo];
      a.h[1] = *(const v8h*)&xs[(nbase + m) * 264 + khi];
      // B fragment (32x16 f16): lane column ccol, 16 consecutive K values
      int kb = kc * 32 + hi * 16;
      v16h b = *(const v16h*)(Wt + (size_t)ccol * kDin + kb);
      acc = __builtin_amdgcn_wmma_f32_16x16x32_f16(false, a.v, false, b,
                                                   (short)0, acc, false, false);
    }

    // store: C/D vgpr r holds (row nbase + hi*8 + r, col ccol)
    int head = ccol >> 6;
    int d    = ccol & 63;
    _Float16* hb = hOut + (size_t)p * kHNd + head * (kN * kD) + d;
    for (int r = 0; r < 8; ++r) {
      int n = nbase + hi * 8 + r;
      hb[(size_t)n * kD] = (_Float16)acc[r];
    }
  }
}

// ---------------------------------------------------------------------------
// Kernel 2: per pair p=(j,i), per head (wave):
//   sims = relu( h[j,i,head] @ h[i,j,head]^T / 8 )   (64x64, K=64)
//   out[p] = sum_head mean(sims) * w2[head]
// B^T fragments == contiguous rows of h[i,j,head] (row-major) -> 32B loads.
// h (64 MiB f16) is L2-resident on MI455X (192 MB L2).
// ---------------------------------------------------------------------------
__global__ __launch_bounds__(256) void pair_score(const _Float16* __restrict__ hBuf,
                                                  const float* __restrict__ w2,
                                                  float* __restrict__ out) {
  const int p  = blockIdx.x;                 // j*32 + i
  const int j  = p >> 5, i = p & 31;
  const int pT = i * kS + j;

  const int lane = threadIdx.x & 31;
  const int wave = threadIdx.x >> 5;         // head index
  const int m    = lane & 15;
  const int hi   = lane >> 4;

  const _Float16* hA = hBuf + (size_t)p  * kHNd + wave * (kN * kD);
  const _Float16* hB = hBuf + (size_t)pT * kHNd + wave * (kN * kD);

  float lsum = 0.f;
  for (int nblk = 0; nblk < 4; ++nblk) {
    AFrag a[2];
    int arow = nblk * 16 + m;
    for (int kc = 0; kc < 2; ++kc) {
      int klo = kc * 32 + hi * 16;
      int khi = kc * 32 + hi * 16 + 8;
      a[kc].h[0] = *(const v8h*)(hA + (size_t)arow * kD + klo);
      a[kc].h[1] = *(const v8h*)(hA + (size_t)arow * kD + khi);
    }
    for (int mblk = 0; mblk < 4; ++mblk) {
      int mcol = mblk * 16 + m;              // output column == row of hB
      v8f acc = {};
      for (int kc = 0; kc < 2; ++kc) {
        int kb = kc * 32 + hi * 16;
        v16h b = *(const v16h*)(hB + (size_t)mcol * kD + kb);
        acc = __builtin_amdgcn_wmma_f32_16x16x32_f16(false, a[kc].v, false, b,
                                                     (short)0, acc, false, false);
      }
      for (int r = 0; r < 8; ++r) lsum += fmaxf(acc[r], 0.f);
    }
  }

  // wave32 reduction
  for (int off = 16; off >= 1; off >>= 1)
    lsum += __shfl_xor(lsum, off, 32);

  __shared__ float sh[kH];
  if (lane == 0) sh[wave] = lsum * w2[wave];
  __syncthreads();
  if (threadIdx.x == 0) {
    float s = 0.f;
    for (int hh = 0; hh < kH; ++hh) s += sh[hh];
    // relu(dot/8) = relu(dot)/8 ; mean over 64*64 -> combined 1/32768
    out[p] = s * (1.0f / 32768.0f);
  }
}

// ---------------------------------------------------------------------------
extern "C" void kernel_launch(void* const* d_in, const int* in_sizes, int n_in,
                              void* d_out, int out_size, void* d_ws, size_t ws_size,
                              hipStream_t stream) {
  const float* x  = (const float*)d_in[0];   // [32,32,64,256] fp32
  const float* W  = (const float*)d_in[1];   // [256,512]      fp32
  const float* w2 = (const float*)d_in[2];   // [8,1]          fp32
  float* out = (float*)d_out;                // [32,32]        fp32

  // workspace layout: Wt f16 (256 KiB) | h f16 (64 MiB)
  _Float16* Wt   = (_Float16*)d_ws;
  _Float16* hBuf = (_Float16*)((char*)d_ws + (size_t)kHD * kDin * sizeof(_Float16));

  wt_convert<<<(kHD * kDin) / 256, 256, 0, stream>>>(W, Wt);
  proj_gemm <<<kS * kS, 256, 0, stream>>>(x, Wt, hBuf);
  pair_score<<<kS * kS, 256, 0, stream>>>(hBuf, w2, out);
}
